// SplitGrid_59966333387114
// MI455X (gfx1250) — matrix operand order repaired and checked by reading, not compiled, not run on hardware
//
#include <hip/hip_runtime.h>

typedef __attribute__((ext_vector_type(16))) _Float16 v16h;
typedef __attribute__((ext_vector_type(8)))  float    v8f;

#define GRID_W 4
#define GRID_H 4
#define NEXP   16
#define IN_DIM 2
#define HID    256
#define OUTD   3

// d_ws int layout:
//   meta[0..15]   : per-expert counts
//   meta[16..32]  : padded exclusive offsets aoff[0..16] (aoff[16] = padded total)
//   meta[48..63]  : scatter cursors
//   meta[64..]    : sorted sample indices (N + padding)

__device__ __forceinline__ int expert_of(float x, float y) {
    int i = (int)fminf(fmaxf(floorf(x * (float)GRID_W), 0.f), (float)(GRID_W - 1));
    int j = (int)fminf(fmaxf(floorf(y * (float)GRID_H), 0.f), (float)(GRID_H - 1));
    return j * GRID_W + i;
}

__global__ void k_zero(int* __restrict__ meta) {
    int t = threadIdx.x;
    if (t < NEXP) {
        meta[t]      = 0;   // counts
        meta[48 + t] = 0;   // cursors
    }
}

__global__ void k_count(const float* __restrict__ samples, int n, int* __restrict__ meta) {
    int idx = blockIdx.x * blockDim.x + threadIdx.x;
    if (idx >= n) return;
    float x = samples[idx * 2 + 0];
    float y = samples[idx * 2 + 1];
    atomicAdd(&meta[expert_of(x, y)], 1);
}

__global__ void k_scan(int* __restrict__ meta) {
    if (threadIdx.x != 0 || blockIdx.x != 0) return;
    int acc = 0;
    for (int e = 0; e < NEXP; ++e) {
        meta[16 + e] = acc;
        int c = meta[e];
        acc += ((c + 15) >> 4) << 4;   // pad each bucket to whole 16-sample tiles
    }
    meta[32] = acc;                    // padded total
}

__global__ void k_scatter(const float* __restrict__ samples, int n,
                          int* __restrict__ meta, int* __restrict__ sorted) {
    int idx = blockIdx.x * blockDim.x + threadIdx.x;
    if (idx >= n) return;
    float x = samples[idx * 2 + 0];
    float y = samples[idx * 2 + 1];
    int e = expert_of(x, y);
    int pos = meta[16 + e] + atomicAdd(&meta[48 + e], 1);
    sorted[pos] = idx;
}

// One wave32 per 16-sample tile of one expert.
// Layer1 (K=2) fused into A-fragment construction; layer2 = 8x v_wmma_f32_16x16x32_f16.
__global__ void __launch_bounds__(32)
k_moe_wmma(const float* __restrict__ samples,
           const float* __restrict__ W1, const float* __restrict__ b1,
           const float* __restrict__ W2, const float* __restrict__ b2,
           const int* __restrict__ meta, const int* __restrict__ sorted,
           float* __restrict__ out) {
    __shared__ float sW1[IN_DIM * HID];   // W1[e] : [2][256]
    __shared__ float sB1[HID];            // b1[e]
    __shared__ float sW2[HID * OUTD];     // W2[e] : [256][3]
    __shared__ float sx[16], sy[16];
    __shared__ int   sIdx[16];

    int t    = blockIdx.x;
    int pos0 = t * 16;
    if (pos0 >= meta[32]) return;         // past padded total (uniform exit)

    // locate expert bucket containing this tile
    int e = 0;
    while (e < NEXP - 1 && meta[16 + e + 1] <= pos0) ++e;
    int cnt        = meta[e];
    int bucketBase = meta[16 + e];
    int nvalid     = bucketBase + cnt - pos0;
    if (nvalid > 16) nvalid = 16;         // >=1 guaranteed by tile construction

    int lane = threadIdx.x;

    const float* gW1 = W1 + (size_t)e * IN_DIM * HID;
    const float* gB1 = b1 + (size_t)e * HID;
    const float* gW2 = W2 + (size_t)e * HID * OUTD;
    __builtin_prefetch(gW2, 0, 1);        // global_prefetch_b8

    for (int i = lane; i < IN_DIM * HID; i += 32) sW1[i] = gW1[i];
    for (int i = lane; i < HID;          i += 32) sB1[i] = gB1[i];
    for (int i = lane; i < HID * OUTD;   i += 32) sW2[i] = gW2[i];

    if (lane < 16) {
        int p   = pos0 + lane;
        int pc  = bucketBase + cnt - 1;
        int sid = sorted[p < pc ? p : pc];  // clamp padding lanes to last valid
        sIdx[lane] = sid;
        sx[lane]   = samples[sid * 2 + 0];
        sy[lane]   = samples[sid * 2 + 1];
    }
    __syncthreads();

    const int m  = lane & 15;   // A row / C column index
    const int hi = lane >> 4;
    const int n  = lane & 15;   // B column (only 0..2 real)
    const float x0 = sx[m];
    const float x1 = sy[m];

    v8f c = {};
    #pragma unroll
    for (int kc = 0; kc < HID / 32; ++kc) {   // 8 chunks of K=32
        v16h a, b;
        const int kbaseA = kc * 32 + hi * 8;
        #pragma unroll
        for (int i = 0; i < 16; ++i) {
            const int v = i >> 1, p = i & 1;
            // A-fragment K pattern (16-bit A 16x32): lanes 0-15 own K{0-7,16-23}(+8 if hi)
            const int kl = (v < 4) ? (2 * v + p) : (16 + 2 * (v - 4) + p);
            const int k  = kbaseA + kl;
            // fused layer 1: h = relu(x @ W1[e] + b1[e])
            float h = fmaf(x0, sW1[k], fmaf(x1, sW1[HID + k], sB1[k]));
            h = fmaxf(h, 0.f);
            a[i] = (_Float16)h;
            // B-fragment: W2[e][k][n], K = kc*32 + same K pattern as A
            const int kb  = kc * 32 + ((i < 8) ? (hi * 8 + i) : (16 + hi * 8 + (i - 8)));
            const int col = (n < OUTD) ? n : 0;
            const float w = sW2[kb * OUTD + col];
            b[i] = (n < OUTD) ? (_Float16)w : (_Float16)0.f;
        }
        c = __builtin_amdgcn_wmma_f32_16x16x32_f16(
                /*neg_a=*/false, a, /*neg_b=*/false, b,
                /*c_mod=*/(short)0, c, /*reuse_a=*/false, /*reuse_b=*/false);
    }

    // C layout: element r of lane L -> row M = r + 8*hi, col N = L&15
    if (n < OUTD) {
        const float bias = b2[e * OUTD + n];
        #pragma unroll
        for (int r = 0; r < 8; ++r) {
            const int row = r + hi * 8;
            if (row < nvalid) {
                out[(size_t)sIdx[row] * OUTD + n] = c[r] + bias;
            }
        }
    }
}

extern "C" void kernel_launch(void* const* d_in, const int* in_sizes, int n_in,
                              void* d_out, int out_size, void* d_ws, size_t ws_size,
                              hipStream_t stream) {
    const float* samples = (const float*)d_in[0];
    const float* W1      = (const float*)d_in[1];
    const float* b1      = (const float*)d_in[2];
    const float* W2      = (const float*)d_in[3];
    const float* b2      = (const float*)d_in[4];
    float*       out     = (float*)d_out;

    const int N = in_sizes[0] / IN_DIM;

    int* meta   = (int*)d_ws;      // 64 ints of counters/offsets
    int* sorted = meta + 64;       // N + padding sample indices

    k_zero<<<1, 32, 0, stream>>>(meta);
    k_count<<<(N + 255) / 256, 256, 0, stream>>>(samples, N, meta);
    k_scan<<<1, 1, 0, stream>>>(meta);
    k_scatter<<<(N + 255) / 256, 256, 0, stream>>>(samples, N, meta, sorted);

    const int maxTiles = (N >> 4) + NEXP;   // upper bound incl. per-bucket padding
    k_moe_wmma<<<maxTiles, 32, 0, stream>>>(samples, W1, b1, W2, b2, meta, sorted, out);
}
